// BaselineGNN_38482906972370
// MI455X (gfx1250) — matrix operand order repaired and checked by reading, not compiled, not run on hardware
//
#include <hip/hip_runtime.h>
#include <hip/hip_bf16.h>
#include <math.h>

typedef __attribute__((ext_vector_type(16))) _Float16 v16h;
typedef __attribute__((ext_vector_type(8)))  _Float16 v8h;
typedef __attribute__((ext_vector_type(8)))  float    v8f;

#define NNODES 50000
#define NEDGES 800000

__device__ __forceinline__ float gelu_exact(float x) {
    // jax.nn.gelu(approximate=False): 0.5*x*(1+erf(x/sqrt(2)))
    return 0.5f * x * (1.0f + erff(x * 0.70710678118654752f));
}

// ---------------------------------------------------------------------------
// Fold inference BatchNorm into the following Dense and emit the weights
// directly in WMMA B-fragment order:
//   fragment element (colTile, kb, lane, j)  <->  B[kb*32 + (j>>3)*16 + (lane>>4)*8 + (j&7)]
//                                                  [colTile*16 + (lane&15)]
// so the GEMM loads one contiguous 32B v16h per wave per k-step.
//   W'[k][n] = (gamma[k]/sqrt(1+eps)) * W[k][n]
//   b'[n]    = b[n] + sum_k beta[k]*W[k][n]
// ---------------------------------------------------------------------------
__global__ void fold_weights_frag(const float* __restrict__ W,
                                  const float* __restrict__ b,
                                  const float* __restrict__ gamma,
                                  const float* __restrict__ beta,
                                  _Float16* __restrict__ Wfrag,
                                  float* __restrict__ bf,
                                  int K, int Nout, int hasBN) {
    int t = blockIdx.x * blockDim.x + threadIdx.x;
    if (t >= K * Nout) return;
    int k = t / Nout;
    int n = t % Nout;
    float s = hasBN ? gamma[k] * rsqrtf(1.0f + 1e-3f) : 1.0f;

    const int KB = K >> 5;              // k-blocks of 32
    const int kb = k >> 5;
    const int r  = k & 31;
    const int lane = (((r >> 3) & 1) << 4) + (n & 15);
    const int j    = (r & 7) + ((r >> 4) << 3);
    const int colTile = n >> 4;
    Wfrag[((size_t)(colTile * KB + kb) * 32 + lane) * 16 + j] =
        (_Float16)(s * W[t]);

    if (k == 0) {
        float acc = b[n];
        if (hasBN) {
            for (int kk = 0; kk < K; ++kk) acc += beta[kk] * W[kk * Nout + n];
        }
        bf[n] = acc;
    }
}

__global__ void zero_f32(float* __restrict__ p, int n) {
    int t = blockIdx.x * blockDim.x + threadIdx.x;
    if (t < n) p[t] = 0.0f;
}

__global__ void f32_to_f16(const float* __restrict__ in,
                           _Float16* __restrict__ out, int n) {
    int t = blockIdx.x * blockDim.x + threadIdx.x;
    if (t < n) out[t] = (_Float16)in[t];
}

__global__ void count_edges(const int* __restrict__ node_idx,
                            float* __restrict__ cnt, int E) {
    int t = blockIdx.x * blockDim.x + threadIdx.x;
    if (t < E) atomicAdd(&cnt[node_idx[t]], 1.0f);
}

// agg[node_idx[e]] += ew[e] * P[nbr_idx[e]]   (F = 4*F4 floats per row)
__global__ void scatter_accum(const float* __restrict__ P,
                              const int* __restrict__ node_idx,
                              const int* __restrict__ nbr_idx,
                              const float* __restrict__ ew,
                              float* __restrict__ agg,
                              int F4, int E) {
    int t = blockIdx.x * blockDim.x + threadIdx.x;
    if (t >= E * F4) return;
    int e = t / F4;
    int q = t - e * F4;
    float w = ew[e];
    const float4 v =
        reinterpret_cast<const float4*>(P + (size_t)nbr_idx[e] * (F4 * 4))[q];
    float* dst = agg + (size_t)node_idx[e] * (F4 * 4) + q * 4;
    atomicAdd(dst + 0, v.x * w);
    atomicAdd(dst + 1, v.y * w);
    atomicAdd(dst + 2, v.z * w);
    atomicAdd(dst + 3, v.w * w);
}

// aggh[i][f] = (half)( agg[i][f] / max(cnt[i], 1) )   -> f16 A-operand
__global__ void normalize_mean_h(const float* __restrict__ agg,
                                 const float* __restrict__ cnt,
                                 _Float16* __restrict__ out,
                                 int F, int N) {
    int t = blockIdx.x * blockDim.x + threadIdx.x;
    if (t >= N * F) return;
    int i = t / F;
    out[t] = (_Float16)(agg[t] / fmaxf(cnt[i], 1.0f));
}

// ---------------------------------------------------------------------------
// Fused split-K-concat GEMM + bias + activation, fully unrolled WMMA f16.
//   Out[M,NOUT] = act( concat_K(A0[:, :KSPLIT], A1[:, :KTOT-KSPLIT]) @ W + b )
// A0/A1: f16 row-major (LDA0/LDA1). Wfrag: pre-packed B fragments.
// One wave -> one 16x16 tile; 8 waves / 256-thread block.
// A fragment: ISA 16-bit A layout (two contiguous 8-half runs per lane).
// ACT: 0 = exact GELU, 1 = ReLU. HALF_OUT: store f16 (A-operand for next
// GEMM) instead of f32.
// ---------------------------------------------------------------------------
template <int KTOT, int KSPLIT, int LDA0, int LDA1, int NOUT, int CT, int RT,
          int ACT, int HALF_OUT>
__global__ void __launch_bounds__(256)
wmma_ffn(const _Float16* __restrict__ A0,
         const _Float16* __restrict__ A1,
         const _Float16* __restrict__ Wfrag,
         const float* __restrict__ bf,
         void* __restrict__ OutRaw,
         int M) {
    constexpr int KB = KTOT / 32;
    const int lane = threadIdx.x & 31;
    const int wave = threadIdx.x >> 5;
    const int colTile = wave % CT;
    const int rowBase = (blockIdx.x * RT + wave / CT) * 16;
    if (rowBase >= M) return;            // wave-uniform: EXEC all-ones at WMMA

    const int g    = lane >> 4;          // half-wave group
    const int l15  = lane & 15;
    const int arow = rowBase + l15;      // A row owned by this lane
    const int col  = colTile * 16 + l15; // B/C/D column owned by this lane

    v8f c = {};
#pragma unroll
    for (int kb = 0; kb < KB; ++kb) {
        constexpr int unused = 0; (void)unused;
        const int k0 = kb * 32;
        const _Float16* src = (k0 < KSPLIT)
            ? (A0 + (size_t)arow * LDA0 + k0)
            : (A1 + (size_t)arow * LDA1 + (k0 - KSPLIT));
        union { v16h v; v8h h[2]; } au;
        au.h[0] = *reinterpret_cast<const v8h*>(src + g * 8);       // K g*8+j
        au.h[1] = *reinterpret_cast<const v8h*>(src + 16 + g * 8);  // K 16+g*8+j
        const v16h b = *reinterpret_cast<const v16h*>(
            Wfrag + ((size_t)(colTile * KB + kb) * 32 + lane) * 16);
        c = __builtin_amdgcn_wmma_f32_16x16x32_f16(
                /*neg_a=*/false, au.v, /*neg_b=*/false, b,
                /*c_mod=*/(short)0, c, /*reuse_a=*/false, /*reuse_b=*/false);
    }

    const float bias = bf[col];
#pragma unroll
    for (int r = 0; r < 8; ++r) {
        const int row = rowBase + g * 8 + r;
        float v = c[r] + bias;
        v = ACT ? fmaxf(v, 0.0f) : gelu_exact(v);
        if (HALF_OUT)
            ((_Float16*)OutRaw)[(size_t)row * NOUT + col] = (_Float16)v;
        else
            ((float*)OutRaw)[(size_t)row * NOUT + col] = v;
    }
}

// out[i] = sigmoid( h[i,:128] . w + b )
__global__ void head_sigmoid(const float* __restrict__ h,
                             const float* __restrict__ w,
                             const float* __restrict__ b,
                             float* __restrict__ out, int N) {
    int i = blockIdx.x * blockDim.x + threadIdx.x;
    if (i >= N) return;
    const float4* hp = reinterpret_cast<const float4*>(h + (size_t)i * 128);
    const float4* wp = reinterpret_cast<const float4*>(w);
    float acc = b[0];
#pragma unroll
    for (int q = 0; q < 32; ++q) {
        float4 a = hp[q], ww = wp[q];
        acc += a.x * ww.x + a.y * ww.y + a.z * ww.z + a.w * ww.w;
    }
    out[i] = 1.0f / (1.0f + expf(-acc));
}

// ---------------------------------------------------------------------------
extern "C" void kernel_launch(void* const* d_in, const int* in_sizes, int n_in,
                              void* d_out, int out_size, void* d_ws,
                              size_t ws_size, hipStream_t stream) {
    (void)in_sizes; (void)n_in; (void)out_size; (void)ws_size;
    const int N = NNODES, E = NEDGES;

    const float* x      = (const float*)d_in[0];
    const int*   edges  = (const int*)  d_in[1];
    const float* ew     = (const float*)d_in[2];
    const int* node_idx = edges;       // edges[0]
    const int* nbr_idx  = edges + E;   // edges[1]

    // workspace carve-out (256B-aligned)
    char* ws = (char*)d_ws;
    size_t off = 0;
    auto carve = [&](size_t bytes) -> void* {
        void* p = ws + off;
        off = (off + bytes + 255) & ~(size_t)255;
        return p;
    };
    float*    cnt   = (float*)   carve((size_t)N * sizeof(float));
    float*    agg1  = (float*)   carve((size_t)N * 64 * sizeof(float));
    float*    agg2  = (float*)   carve((size_t)N * 32 * sizeof(float));
    float*    P1    = (float*)   carve((size_t)N * 64 * sizeof(float));
    float*    P2    = (float*)   carve((size_t)N * 32 * sizeof(float));
    float*    hbuf  = (float*)   carve((size_t)N * 128 * sizeof(float));
    _Float16* xh    = (_Float16*)carve((size_t)N * 128 * sizeof(_Float16));
    _Float16* x1h   = (_Float16*)carve((size_t)N * 64 * sizeof(_Float16));
    _Float16* x2h   = (_Float16*)carve((size_t)N * 32 * sizeof(_Float16));
    _Float16* agg1h = (_Float16*)carve((size_t)N * 64 * sizeof(_Float16));
    _Float16* agg2h = (_Float16*)carve((size_t)N * 32 * sizeof(_Float16));
    _Float16* W1p   = (_Float16*)carve(128 * 64 * sizeof(_Float16));
    _Float16* W1u   = (_Float16*)carve(192 * 64 * sizeof(_Float16));
    _Float16* W2p   = (_Float16*)carve( 64 * 32 * sizeof(_Float16));
    _Float16* W2u   = (_Float16*)carve( 96 * 32 * sizeof(_Float16));
    _Float16* Wd1   = (_Float16*)carve( 32 * 128 * sizeof(_Float16));
    float*    b1p   = (float*)   carve( 64 * sizeof(float));
    float*    b1u   = (float*)   carve( 64 * sizeof(float));
    float*    b2p   = (float*)   carve( 32 * sizeof(float));
    float*    b2u   = (float*)   carve( 32 * sizeof(float));
    float*    bd1   = (float*)   carve(128 * sizeof(float));

    const int T = 256;
    auto blocks = [](long long n, int t) { return (int)((n + t - 1) / t); };

    // --- fold BN into Dense weights, pre-packed B fragments ------------------
    fold_weights_frag<<<blocks(128 * 64, T), T, 0, stream>>>(
        (const float*)d_in[5], (const float*)d_in[6],
        (const float*)d_in[3], (const float*)d_in[4], W1p, b1p, 128, 64, 1);
    fold_weights_frag<<<blocks(192 * 64, T), T, 0, stream>>>(
        (const float*)d_in[9], (const float*)d_in[10],
        (const float*)d_in[7], (const float*)d_in[8], W1u, b1u, 192, 64, 1);
    fold_weights_frag<<<blocks(64 * 32, T), T, 0, stream>>>(
        (const float*)d_in[13], (const float*)d_in[14],
        (const float*)d_in[11], (const float*)d_in[12], W2p, b2p, 64, 32, 1);
    fold_weights_frag<<<blocks(96 * 32, T), T, 0, stream>>>(
        (const float*)d_in[17], (const float*)d_in[18],
        (const float*)d_in[15], (const float*)d_in[16], W2u, b2u, 96, 32, 1);
    fold_weights_frag<<<blocks(32 * 128, T), T, 0, stream>>>(
        (const float*)d_in[19], (const float*)d_in[20],
        (const float*)d_in[20], (const float*)d_in[20], Wd1, bd1, 32, 128, 0);

    // --- f16 copy of node features (A-operand of GEMM1/GEMM2) ----------------
    f32_to_f16<<<blocks((long long)N * 128, T), T, 0, stream>>>(
        x, xh, N * 128);

    // --- zero accumulators (re-done every call: deterministic) ---------------
    zero_f32<<<blocks(N, T), T, 0, stream>>>(cnt, N);
    zero_f32<<<blocks((long long)N * 64, T), T, 0, stream>>>(agg1, N * 64);
    zero_f32<<<blocks((long long)N * 32, T), T, 0, stream>>>(agg2, N * 32);

    count_edges<<<blocks(E, T), T, 0, stream>>>(node_idx, cnt, E);

    // --- layer 1: prep FFN (128->64), edge mean, update (concat 192->64) -----
    wmma_ffn<128, 128, 128, 128, 64, 4, 2, 0, 0>
        <<<blocks(N / 16, 2), T, 0, stream>>>(xh, xh, W1p, b1p, P1, N);
    scatter_accum<<<blocks((long long)E * 16, T), T, 0, stream>>>(
        P1, node_idx, nbr_idx, ew, agg1, 16, E);
    normalize_mean_h<<<blocks((long long)N * 64, T), T, 0, stream>>>(
        agg1, cnt, agg1h, 64, N);
    wmma_ffn<192, 128, 128, 64, 64, 4, 2, 0, 1>
        <<<blocks(N / 16, 2), T, 0, stream>>>(xh, agg1h, W1u, b1u, x1h, N);

    // --- layer 2: prep (64->32), edge mean, update (concat 96->32) -----------
    wmma_ffn<64, 64, 64, 64, 32, 2, 4, 0, 0>
        <<<blocks(N / 16, 4), T, 0, stream>>>(x1h, x1h, W2p, b2p, P2, N);
    scatter_accum<<<blocks((long long)E * 8, T), T, 0, stream>>>(
        P2, node_idx, nbr_idx, ew, agg2, 8, E);
    normalize_mean_h<<<blocks((long long)N * 32, T), T, 0, stream>>>(
        agg2, cnt, agg2h, 32, N);
    wmma_ffn<96, 64, 64, 32, 32, 2, 4, 0, 1>
        <<<blocks(N / 16, 4), T, 0, stream>>>(x1h, agg2h, W2u, b2u, x2h, N);

    // --- head: Dense 32->128 + ReLU (WMMA), then 128->1 + sigmoid ------------
    wmma_ffn<32, 32, 32, 32, 128, 8, 1, 1, 0>
        <<<N / 16, T, 0, stream>>>(x2h, x2h, Wd1, bd1, hbuf, N);
    head_sigmoid<<<blocks(N, T), T, 0, stream>>>(
        hbuf, (const float*)d_in[21], (const float*)d_in[22],
        (float*)d_out, N);
}
